// Model_4569845202964
// MI455X (gfx1250) — compile-verified, matrix operand order checked by reading, not run on hardware
//
#include <hip/hip_runtime.h>
#include <math.h>

typedef __bf16 bf16;
typedef __attribute__((ext_vector_type(16))) __bf16 v16bf;
typedef __attribute__((ext_vector_type(2)))  __bf16 bf16x2;
typedef __attribute__((ext_vector_type(8)))  float  v8f;

#define NTOK 866
#define BDIM 8
#define LSEQ 96
#define CIN  862
#define DMOD 512
#define DINN 1024
#define DST  16
#define DTRK 32

#define BM 64
#define BN 64
#define BK 32
#define MROWS (BDIM * NTOK)                 /* 6928 */
#define MPAD  (((MROWS + BM - 1) / BM) * BM) /* 6976 */

// ---------------------------------------------------------------------------
// Stats + token build: tok[b*866+c, t] = (x_enc[b,t,c]-mean)/std  (c<862)
//                                       = x_mark[b,t,c-862]        (c>=862)
// ---------------------------------------------------------------------------
__global__ __launch_bounds__(128) void k_tok(const float* __restrict__ xe,
                                             const float* __restrict__ xm,
                                             float* __restrict__ tok,
                                             float* __restrict__ means,
                                             float* __restrict__ stdev)
{
    int row = blockIdx.x;                // b*866 + c
    int b = row / NTOK, c = row - b * NTOK;
    int t = threadIdx.x;
    __shared__ float s1[128], s2[128];
    if (c < CIN) {
        float v = 0.f;
        if (t < LSEQ) v = xe[(size_t)b * LSEQ * CIN + (size_t)t * CIN + c];
        s1[t] = v; s2[t] = v * v;
        __syncthreads();
        #pragma unroll
        for (int o = 64; o > 0; o >>= 1) {
            if (t < o) { s1[t] += s1[t + o]; s2[t] += s2[t + o]; }
            __syncthreads();
        }
        float mean = s1[0] * (1.f / 96.f);
        float var  = s2[0] * (1.f / 96.f) - mean * mean;
        float sd   = sqrtf(var + 1e-5f);
        if (t == 0) { means[b * CIN + c] = mean; stdev[b * CIN + c] = sd; }
        if (t < LSEQ) tok[(size_t)row * LSEQ + t] = (v - mean) / sd;
    } else {
        if (t < LSEQ) tok[(size_t)row * LSEQ + t] =
            xm[(size_t)b * LSEQ * 4 + (size_t)t * 4 + (c - CIN)];
    }
}

// ---------------------------------------------------------------------------
// bf16 WMMA GEMM:  C[M,N] = A[M,K] @ W[N,K]^T (+bias) (+softplus)
// A is a padded workspace buffer (>= gridDim.x*BM rows) -> unguarded float4
// staging. W rows are clamped to N-1 (unconditional loads, guarded stores).
// Block: 256 thr = 8 waves as 4x2; each wave: 1 A-frag, 2 B-frags, 2 WMMAs.
// act: 0 = none, 1 = softplus
// ---------------------------------------------------------------------------
__global__ __launch_bounds__(256) void wmma_gemm(const float* __restrict__ A, int lda,
                                                 const float* __restrict__ W,
                                                 const float* __restrict__ bias,
                                                 float* __restrict__ C, int ldc,
                                                 int M, int N, int K, int act)
{
    __shared__ bf16x2 As2[BM][BK / 2 + 1];   // stride 17 dwords
    __shared__ bf16x2 Ws2[BN][BK / 2 + 1];

    const int tid  = threadIdx.x;
    const int lane = tid & 31;
    const int wid  = tid >> 5;
    const int wm   = wid & 3;           // 4 wave-rows of 16
    const int wn   = wid >> 2;          // 2 wave-cols of 32
    const int bm0  = blockIdx.x * BM;
    const int bn0  = blockIdx.y * BN;

    const int l15 = lane & 15;
    const int kb2 = (lane < 16) ? 0 : 4;    // dword k-base (k=0/8 in halves)
    const int ar  = wm * 16 + l15;          // A row within tile
    const int bc0 = wn * 32 + l15;          // first B col within tile
    const int bc1 = bc0 + 16;

    // staging coords: 64 rows x 32 k, 8 f32 per thread (two float4)
    const int sr = tid >> 2;            // 0..63
    const int sk = (tid & 3) * 8;       // 0,8,16,24
    const int gr = bm0 + sr;                      // unguarded (padded ws)
    const int gn = min(bn0 + sr, N - 1);          // clamped weight row

    v8f acc0 = {}, acc1 = {};

    for (int k0 = 0; k0 < K; k0 += BK) {
        const float4 a0 = *(const float4*)(A + (size_t)gr * lda + k0 + sk);
        const float4 a1 = *(const float4*)(A + (size_t)gr * lda + k0 + sk + 4);
        const float4 w0 = *(const float4*)(W + (size_t)gn * K + k0 + sk);
        const float4 w1 = *(const float4*)(W + (size_t)gn * K + k0 + sk + 4);
        if (k0 + BK < K) {
            __builtin_prefetch(A + (size_t)gr * lda + k0 + BK + sk, 0, 3);
            __builtin_prefetch(W + (size_t)gn * K + k0 + BK + sk, 0, 3);
        }
        const int kd = sk >> 1;
        As2[sr][kd + 0] = bf16x2{(bf16)a0.x, (bf16)a0.y};
        As2[sr][kd + 1] = bf16x2{(bf16)a0.z, (bf16)a0.w};
        As2[sr][kd + 2] = bf16x2{(bf16)a1.x, (bf16)a1.y};
        As2[sr][kd + 3] = bf16x2{(bf16)a1.z, (bf16)a1.w};
        Ws2[sr][kd + 0] = bf16x2{(bf16)w0.x, (bf16)w0.y};
        Ws2[sr][kd + 1] = bf16x2{(bf16)w0.z, (bf16)w0.w};
        Ws2[sr][kd + 2] = bf16x2{(bf16)w1.x, (bf16)w1.y};
        Ws2[sr][kd + 3] = bf16x2{(bf16)w1.z, (bf16)w1.w};
        __syncthreads();

        // fragments per documented CDNA5 16-bit layout
        v16bf af, bf0, bf1;
        #pragma unroll
        for (int j = 0; j < 4; ++j) {
            bf16x2 p0 = As2[ar][kb2 + j];
            bf16x2 p1 = As2[ar][kb2 + 8 + j];
            af[2 * j]     = p0[0];  af[2 * j + 1]     = p0[1];
            af[8 + 2 * j] = p1[0];  af[8 + 2 * j + 1] = p1[1];
            bf16x2 q0 = Ws2[bc0][kb2 + j];
            bf16x2 q1 = Ws2[bc0][kb2 + 8 + j];
            bf0[2 * j]     = q0[0];  bf0[2 * j + 1]     = q0[1];
            bf0[8 + 2 * j] = q1[0];  bf0[8 + 2 * j + 1] = q1[1];
            bf16x2 r0 = Ws2[bc1][kb2 + j];
            bf16x2 r1 = Ws2[bc1][kb2 + 8 + j];
            bf1[2 * j]     = r0[0];  bf1[2 * j + 1]     = r0[1];
            bf1[8 + 2 * j] = r1[0];  bf1[8 + 2 * j + 1] = r1[1];
        }

        acc0 = __builtin_amdgcn_wmma_f32_16x16x32_bf16(
                   false, af, false, bf0, (short)0, acc0, false, false);
        acc1 = __builtin_amdgcn_wmma_f32_16x16x32_bf16(
                   false, af, false, bf1, (short)0, acc1, false, false);
        __syncthreads();
    }

    // epilogue: C/D layout VGPR i -> row (i | i+8), lane -> col
    const int col0  = bn0 + bc0;
    const int col1  = bn0 + bc1;
    const int mbase = bm0 + wm * 16 + ((lane < 16) ? 0 : 8);
    const float bv0 = (bias != nullptr && col0 < N) ? bias[col0] : 0.f;
    const float bv1 = (bias != nullptr && col1 < N) ? bias[col1] : 0.f;
    #pragma unroll
    for (int i = 0; i < 8; ++i) {
        int m = mbase + i;
        if (m < M) {
            if (col0 < N) {
                float v = acc0[i] + bv0;
                if (act == 1) v = (v > 20.f) ? v : log1pf(__expf(v));
                C[(size_t)m * ldc + col0] = v;
            }
            if (col1 < N) {
                float v = acc1[i] + bv1;
                if (act == 1) v = (v > 20.f) ? v : log1pf(__expf(v));
                C[(size_t)m * ldc + col1] = v;
            }
        }
    }
}

// ---------------------------------------------------------------------------
// Depthwise causal conv (DC=4) over token axis + bias + SiLU.
// u = xz[:, :1024] with row stride 2048.
// ---------------------------------------------------------------------------
__global__ __launch_bounds__(256) void k_conv(const float* __restrict__ xz,
                                              const float* __restrict__ cw,
                                              const float* __restrict__ cb,
                                              float* __restrict__ uc, int total)
{
    int idx = blockIdx.x * 256 + threadIdx.x;
    if (idx >= total) return;
    int d = idx & 1023;
    int r = idx >> 10;                 // b*866 + t
    int t = r % NTOK;
    float acc = cb[d];
    #pragma unroll
    for (int k = 0; k < 4; ++k) {
        int off = k - 3;               // causal tap
        if (t + off >= 0)
            acc += xz[(size_t)(r + off) * 2048 + d] * cw[d * 4 + k];
    }
    uc[(size_t)r * 1024 + d] = acc / (1.f + __expf(-acc));
}

// ---------------------------------------------------------------------------
// Selective scan + D skip + SiLU(z) gate, fused.
// One lane per (b,d); 16 states + 16 A coeffs live in VGPRs.
// B_t/C_t (shared per batch) staged into LDS in 64-step chunks.
// ---------------------------------------------------------------------------
__global__ __launch_bounds__(256) void k_scan(const float* __restrict__ xdbl,
                                              const float* __restrict__ dtb,
                                              const float* __restrict__ uc,
                                              const float* __restrict__ xz,
                                              const float* __restrict__ A_log,
                                              const float* __restrict__ Dp,
                                              float* __restrict__ yg)
{
    int b = blockIdx.x >> 2;
    int d = ((blockIdx.x & 3) << 8) + threadIdx.x;

    float a[16], h[16];
    #pragma unroll
    for (int n = 0; n < 16; ++n) {
        a[n] = -__expf(A_log[d * 16 + n]);
        h[n] = 0.f;
    }
    float Dd = Dp[d];

    __shared__ float bc[64 * 32];      // per step: 16 B then 16 C
    int rowb = b * NTOK;

    for (int t0 = 0; t0 < NTOK; t0 += 64) {
        int Tc = min(64, NTOK - t0);
        __syncthreads();
        for (int idx = threadIdx.x; idx < Tc * 32; idx += 256) {
            int tt = idx >> 5, j = idx & 31;
            bc[idx] = xdbl[(size_t)(rowb + t0 + tt) * 64 + 32 + j];
        }
        __syncthreads();
        for (int tt = 0; tt < Tc; ++tt) {
            size_t row = (size_t)(rowb + t0 + tt);
            float dv = dtb[row * 1024 + d];
            float uv = uc[row * 1024 + d];
            float du = dv * uv;
            float y = 0.f;
            #pragma unroll
            for (int n = 0; n < 16; ++n) {
                float dA = __expf(dv * a[n]);
                h[n] = dA * h[n] + du * bc[tt * 32 + n];
                y += h[n] * bc[tt * 32 + 16 + n];
            }
            float zv = xz[row * 2048 + 1024 + d];
            float g  = zv / (1.f + __expf(-zv));
            yg[row * 1024 + d] = (y + uv * Dd) * g;
        }
    }
}

// ---------------------------------------------------------------------------
// LayerNorm over DM=512 per row.
// ---------------------------------------------------------------------------
__global__ __launch_bounds__(128) void k_ln(const float* __restrict__ hsrc,
                                            const float* __restrict__ g,
                                            const float* __restrict__ bb,
                                            float* __restrict__ hn)
{
    int row = blockIdx.x;
    int t = threadIdx.x;
    float v[4]; float s = 0.f, s2 = 0.f;
    #pragma unroll
    for (int e = 0; e < 4; ++e) {
        v[e] = hsrc[(size_t)row * 512 + t + e * 128];
        s += v[e]; s2 += v[e] * v[e];
    }
    __shared__ float r1[128], r2[128];
    r1[t] = s; r2[t] = s2; __syncthreads();
    #pragma unroll
    for (int o = 64; o > 0; o >>= 1) {
        if (t < o) { r1[t] += r1[t + o]; r2[t] += r2[t + o]; }
        __syncthreads();
    }
    float mu  = r1[0] * (1.f / 512.f);
    float var = r2[0] * (1.f / 512.f) - mu * mu;
    float rs  = rsqrtf(var + 1e-5f);
    #pragma unroll
    for (int e = 0; e < 4; ++e) {
        int c = t + e * 128;
        hn[(size_t)row * 512 + c] = (v[e] - mu) * rs * g[c] + bb[c];
    }
}

// ---------------------------------------------------------------------------
// Transpose + de-normalize: out[b,t,c] = proj[b*866+c, t]*std + mean
// ---------------------------------------------------------------------------
__global__ __launch_bounds__(256) void k_out(const float* __restrict__ pj,
                                             const float* __restrict__ stdev,
                                             const float* __restrict__ means,
                                             float* __restrict__ out, int total)
{
    int idx = blockIdx.x * 256 + threadIdx.x;
    if (idx >= total) return;
    int c = idx % CIN;
    int t = (idx / CIN) % LSEQ;
    int b = idx / (CIN * LSEQ);
    out[idx] = pj[(size_t)(b * NTOK + c) * LSEQ + t] * stdev[b * CIN + c]
             + means[b * CIN + c];
}

// ---------------------------------------------------------------------------
extern "C" void kernel_launch(void* const* d_in, const int* in_sizes, int n_in,
                              void* d_out, int out_size, void* d_ws, size_t ws_size,
                              hipStream_t stream)
{
    const float* x_enc     = (const float*)d_in[0];
    const float* x_mark    = (const float*)d_in[1];
    const float* emb_w     = (const float*)d_in[4];
    const float* emb_b     = (const float*)d_in[5];
    const float* in_proj_w = (const float*)d_in[6];
    const float* conv_w    = (const float*)d_in[7];
    const float* conv_b    = (const float*)d_in[8];
    const float* x_proj_w  = (const float*)d_in[9];
    const float* dt_proj_w = (const float*)d_in[10];
    const float* dt_proj_b = (const float*)d_in[11];
    const float* A_log     = (const float*)d_in[12];
    const float* Dp        = (const float*)d_in[13];
    const float* out_proj_w= (const float*)d_in[14];
    const float* ln_g      = (const float*)d_in[15];
    const float* ln_b      = (const float*)d_in[16];
    const float* proj_w    = (const float*)d_in[17];
    const float* proj_b    = (const float*)d_in[18];
    float* out = (float*)d_out;

    const int M = MROWS;                   // 6928 valid rows
    float* ws = (float*)d_ws;
    size_t off = 0;
    auto alloc = [&](size_t n) { float* p = ws + off; off += (n + 63) & ~(size_t)63; return p; };

    // all row-dim buffers padded to MPAD rows for unguarded GEMM staging
    float* tok   = alloc((size_t)MPAD * LSEQ);
    float* means = alloc((size_t)BDIM * CIN);
    float* stdev = alloc((size_t)BDIM * CIN);
    float* hA    = alloc((size_t)MPAD * DMOD);
    float* hB    = alloc((size_t)MPAD * DMOD);
    float* xz    = alloc((size_t)MPAD * 2 * DINN);
    float* uc    = alloc((size_t)MPAD * DINN);
    float* xdbl  = alloc((size_t)MPAD * 64);
    float* dtb   = alloc((size_t)MPAD * DINN);
    float* yg    = alloc((size_t)MPAD * DINN);
    float* hn    = alloc((size_t)MPAD * DMOD);
    float* pj    = alloc((size_t)MPAD * LSEQ);

    auto gemm = [&](const float* A, int lda, const float* W, const float* bias,
                    float* C, int ldc, int N, int K, int act) {
        dim3 g((M + BM - 1) / BM, (N + BN - 1) / BN);
        wmma_gemm<<<g, 256, 0, stream>>>(A, lda, W, bias, C, ldc, M, N, K, act);
    };

    // 1. normalize + tokenize
    k_tok<<<dim3(M), dim3(128), 0, stream>>>(x_enc, x_mark, tok, means, stdev);

    // 2. embedding: (M,96) x (512,96)^T + b
    gemm(tok, LSEQ, emb_w, emb_b, hA, DMOD, DMOD, LSEQ, 0);

    float* hin = hA; float* hout = hB;
    for (int l = 0; l < 2; ++l) {
        // in_proj: (M,512) x (2048,512)^T
        gemm(hin, DMOD, in_proj_w + (size_t)l * 2 * DINN * DMOD, nullptr,
             xz, 2 * DINN, 2 * DINN, DMOD, 0);
        // depthwise conv + SiLU
        int totc = M * DINN;
        k_conv<<<dim3((totc + 255) / 256), 256, 0, stream>>>(
            xz, conv_w + (size_t)l * DINN * 4, conv_b + (size_t)l * DINN, uc, totc);
        // x_proj: (M,1024) x (64,1024)^T
        gemm(uc, DINN, x_proj_w + (size_t)l * 64 * DINN, nullptr, xdbl, 64, 64, DINN, 0);
        // dt_proj + softplus: (M,32 [stride 64]) x (1024,32)^T + b
        gemm(xdbl, 64, dt_proj_w + (size_t)l * DINN * DTRK,
             dt_proj_b + (size_t)l * DINN, dtb, DINN, DINN, DTRK, 1);
        // fused selective scan + gate
        k_scan<<<dim3(32), 256, 0, stream>>>(
            xdbl, dtb, uc, xz,
            A_log + (size_t)l * DINN * DST, Dp + (size_t)l * DINN, yg);
        // out_proj: (M,1024) x (512,1024)^T
        gemm(yg, DINN, out_proj_w + (size_t)l * DMOD * DINN, nullptr,
             hout, DMOD, DMOD, DINN, 0);
        float* tmp = hin; hin = hout; hout = tmp;
    }

    // layernorm + projection + de-normalized output
    k_ln<<<dim3(M), 128, 0, stream>>>(hin, ln_g, ln_b, hn);
    gemm(hn, DMOD, proj_w, proj_b, pj, LSEQ, LSEQ, DMOD, 0);
    int tot = BDIM * LSEQ * CIN;
    k_out<<<dim3((tot + 255) / 256), 256, 0, stream>>>(pj, stdev, means, out, tot);
    (void)in_sizes; (void)n_in; (void)out_size; (void)ws_size;
}
